// ContrastLayer_2911987826805
// MI455X (gfx1250) — matrix operand order repaired and checked by reading, not compiled, not run on hardware
//
#include <hip/hip_runtime.h>
#include <hip/hip_bf16.h>

#define DIN 128
#define HHD 128   // H*D
#define NH  8
#define HD  16
#define NEG_SLOPE 0.2f

typedef __attribute__((ext_vector_type(16))) __bf16 v16bf;
typedef __attribute__((ext_vector_type(8)))  float  v8f;

__device__ __forceinline__ unsigned short f2bf(float f) {
  // round-to-nearest-even f32 -> bf16
  unsigned int u = __float_as_uint(f);
  unsigned int r = u + 0x7FFFu + ((u >> 16) & 1u);
  return (unsigned short)(r >> 16);
}

__device__ __forceinline__ float lrelu(float v) {
  return v > 0.f ? v : NEG_SLOPE * v;
}

// float atomic max via sign-split integer atomics (order-preserving encodings)
__device__ __forceinline__ void atomicMaxFloat(float* addr, float value) {
  if (value >= 0.f)
    atomicMax((int*)addr, __float_as_int(value));
  else
    atomicMin((unsigned int*)addr, __float_as_uint(value));
}

// -------------------------------------------------------------------------
// K1: h = [feat_src; feat_dst] @ W  using v_wmma_f32_16x16x32_bf16.
// Each wave computes a 16-row x 128-col strip.
// W is staged in LDS PRE-PACKED in the per-lane WMMA B-operand layout:
//   Wpack[(step*8 + nt)*32*16 + lane*16 + j] = bf16( W[(step*32 + (j>>3)*16 + (lane>>4)*8 + (j&7))*128
//                                                      + nt*16 + (lane&15)] )
// so each B operand is two contiguous ds_load_b128 per lane.
// -------------------------------------------------------------------------
__global__ __launch_bounds__(256)
void gat_gemm_wmma(const float* __restrict__ fsrc, const float* __restrict__ fdst,
                   const float* __restrict__ W, float* __restrict__ h,
                   int n_src, int n_total) {
  __shared__ __align__(32) unsigned short Wpack[DIN * HHD]; // 32KB packed bf16 weights

  for (int p = threadIdx.x; p < DIN * HHD; p += blockDim.x) {
    const int j    = p & 15;
    const int ln   = (p >> 4) & 31;
    const int nt   = (p >> 9) & 7;
    const int step = p >> 12;
    const int k    = step * 32 + (j >> 3) * 16 + (ln >> 4) * 8 + (j & 7);
    const int col  = nt * 16 + (ln & 15);
    Wpack[p] = f2bf(W[k * HHD + col]);
  }
  __syncthreads();

  const int lane    = threadIdx.x & 31;
  const int wave    = threadIdx.x >> 5;
  const int rowBase = (blockIdx.x * 8 + wave) * 16;
  const int half    = lane >> 4;   // 0: lanes 0-15, 1: lanes 16-31
  const int l15     = lane & 15;
  // Out-of-range A rows only affect out-of-range D rows (never stored),
  // so clamp instead of zero-fill: unconditional vector loads, EXEC all-ones.
  int row = rowBase + l15;
  if (row > n_total - 1) row = n_total - 1;
  const float* xrow = (row < n_src) ? (fsrc + (size_t)row * DIN)
                                    : (fdst + (size_t)(row - n_src) * DIN);

  v8f acc[8] = {};  // 8 N-tiles of 16x16 f32

  #pragma unroll
  for (int kk = 0; kk < DIN; kk += 32) {
    // A tile 16x32 bf16, ISA layout:
    //  lanes 0-15 : elems 0..7 = K kk+0..7,   elems 8..15 = K kk+16..23
    //  lanes 16-31: elems 0..7 = K kk+8..15,  elems 8..15 = K kk+24..31
    union { v16bf v; unsigned short u[16]; } a;
    const float4* p0 = (const float4*)(xrow + kk + half * 8);
    const float4* p1 = (const float4*)(xrow + kk + 16 + half * 8);
    const float4 f0 = p0[0], f1 = p0[1];
    const float4 g0 = p1[0], g1 = p1[1];
    a.u[0]  = f2bf(f0.x); a.u[1]  = f2bf(f0.y); a.u[2]  = f2bf(f0.z); a.u[3]  = f2bf(f0.w);
    a.u[4]  = f2bf(f1.x); a.u[5]  = f2bf(f1.y); a.u[6]  = f2bf(f1.z); a.u[7]  = f2bf(f1.w);
    a.u[8]  = f2bf(g0.x); a.u[9]  = f2bf(g0.y); a.u[10] = f2bf(g0.z); a.u[11] = f2bf(g0.w);
    a.u[12] = f2bf(g1.x); a.u[13] = f2bf(g1.y); a.u[14] = f2bf(g1.z); a.u[15] = f2bf(g1.w);

    const int stepBase = (kk >> 5) * (8 * 32 * 16) + lane * 16;
    #pragma unroll
    for (int nt = 0; nt < 8; ++nt) {
      union { v16bf v; uint4 q[2]; } b;
      const uint4* bp = (const uint4*)&Wpack[stepBase + nt * (32 * 16)];
      b.q[0] = bp[0];
      b.q[1] = bp[1];
      acc[nt] = __builtin_amdgcn_wmma_f32_16x16x32_bf16(
          false, a.v, false, b.v, (short)0, acc[nt], false, false);
    }
  }

  // C/D layout: elem r -> (M = r + half*8, N = l15) within tile
  #pragma unroll
  for (int nt = 0; nt < 8; ++nt) {
    #pragma unroll
    for (int r = 0; r < 8; ++r) {
      const int orow = rowBase + r + half * 8;
      if (orow < n_total)
        h[(size_t)orow * HHD + nt * 16 + l15] = acc[nt][r];
    }
  }
}

// -------------------------------------------------------------------------
// K2: el[n,h] = <h[n,h,:], attn_l[h,:]> ; er likewise.
// -------------------------------------------------------------------------
__global__ __launch_bounds__(256)
void gat_elr(const float* __restrict__ h, const float* __restrict__ attn_l,
             const float* __restrict__ attn_r, float* __restrict__ el,
             float* __restrict__ er, int n_total) {
  int idx = blockIdx.x * blockDim.x + threadIdx.x;
  if (idx >= n_total * NH) return;
  const int hd = idx & (NH - 1);
  const int n  = idx >> 3;
  const float4* hp = (const float4*)(h + (size_t)n * HHD + hd * HD);
  const float4* al = (const float4*)(attn_l + hd * HD);
  const float4* ar = (const float4*)(attn_r + hd * HD);
  float sl = 0.f, sr = 0.f;
  #pragma unroll
  for (int q = 0; q < 4; ++q) {
    const float4 hv = hp[q];
    const float4 av = al[q];
    const float4 rv = ar[q];
    sl += hv.x * av.x + hv.y * av.y + hv.z * av.z + hv.w * av.w;
    sr += hv.x * rv.x + hv.y * rv.y + hv.z * rv.z + hv.w * rv.w;
  }
  el[idx] = sl;
  er[idx] = sr;
}

// -------------------------------------------------------------------------
// K3a: seed segment max with the self-loop logit (every dst has a self loop).
// -------------------------------------------------------------------------
__global__ __launch_bounds__(256)
void gat_minit(const float* __restrict__ el, const float* __restrict__ er,
               float* __restrict__ m, int n_src, int n_dst) {
  int idx = blockIdx.x * blockDim.x + threadIdx.x;
  if (idx >= n_dst * NH) return;
  const int hd = idx & (NH - 1);
  const int g  = n_src + (idx >> 3);
  m[idx] = lrelu(el[g * NH + hd] + er[g * NH + hd]);
}

// -------------------------------------------------------------------------
// K3b: segment max over real edges (thread per edge*head).
// -------------------------------------------------------------------------
__global__ __launch_bounds__(256)
void gat_emax(const int* __restrict__ esg, const int* __restrict__ edg,
              const float* __restrict__ el, const float* __restrict__ er,
              float* __restrict__ m, int n_src, int nE) {
  int idx = blockIdx.x * blockDim.x + threadIdx.x;
  if (idx >= nE * NH) return;
  const int e  = idx >> 3;
  const int hd = idx & (NH - 1);
  const int s  = esg[e];
  const int d  = edg[e];
  const float v = lrelu(el[s * NH + hd] + er[(n_src + d) * NH + hd]);
  atomicMaxFloat(&m[d * NH + hd], v);
}

// -------------------------------------------------------------------------
// K4: seed z and out with self-loop contribution (unnormalized).
// -------------------------------------------------------------------------
__global__ __launch_bounds__(256)
void gat_init(const float* __restrict__ h, const float* __restrict__ el,
              const float* __restrict__ er, const float* __restrict__ m,
              float* __restrict__ z, float* __restrict__ out,
              int n_src, int n_dst) {
  int idx = blockIdx.x * blockDim.x + threadIdx.x;
  if (idx >= n_dst * HHD) return;
  const int i  = idx >> 7;
  const int c  = idx & (HHD - 1);
  const int hd = c >> 4;
  const int g  = n_src + i;
  const float es = lrelu(el[g * NH + hd] + er[g * NH + hd]);
  const float ee = __expf(es - m[i * NH + hd]);
  if ((c & 15) == 0) z[i * NH + hd] = ee;
  out[idx] = ee * h[(size_t)g * HHD + c];
}

// -------------------------------------------------------------------------
// K5: one wave32 per edge: accumulate z[d,h] and out[d,:] += ee * h[s,:].
// -------------------------------------------------------------------------
__global__ __launch_bounds__(256)
void gat_eacc(const int* __restrict__ esg, const int* __restrict__ edg,
              const float* __restrict__ h, const float* __restrict__ el,
              const float* __restrict__ er, const float* __restrict__ m,
              float* __restrict__ z, float* __restrict__ out,
              int n_src, int nE) {
  const int gid  = blockIdx.x * blockDim.x + threadIdx.x;
  const int lane = threadIdx.x & 31;
  const int e    = gid >> 5;          // uniform per wave
  if (e >= nE) return;
  const int s  = esg[e];
  const int d  = edg[e];
  const int hd = lane & (NH - 1);     // lanes 0-7 own heads 0-7
  const float v   = lrelu(el[s * NH + hd] + er[(n_src + d) * NH + hd]);
  const float eeL = __expf(v - m[d * NH + hd]);
  if (lane < NH) atomicAdd(&z[d * NH + lane], eeL);

  const float* hs = h   + (size_t)s * HHD;
  float*       od = out + (size_t)d * HHD;
  #pragma unroll
  for (int q = 0; q < 4; ++q) {
    const int c = lane + q * 32;
    const float ee = __shfl(eeL, c >> 4, 32);
    atomicAdd(&od[c], ee * hs[c]);
  }
}

// -------------------------------------------------------------------------
// K6: normalize by z and add bias.
// -------------------------------------------------------------------------
__global__ __launch_bounds__(256)
void gat_final(const float* __restrict__ z, const float* __restrict__ bias,
               float* __restrict__ out, int n_dst) {
  int idx = blockIdx.x * blockDim.x + threadIdx.x;
  if (idx >= n_dst * HHD) return;
  const int i  = idx >> 7;
  const int c  = idx & (HHD - 1);
  const int hd = c >> 4;
  out[idx] = out[idx] / z[i * NH + hd] + bias[c];
}

extern "C" void kernel_launch(void* const* d_in, const int* in_sizes, int n_in,
                              void* d_out, int out_size, void* d_ws, size_t ws_size,
                              hipStream_t stream) {
  const float* fsrc   = (const float*)d_in[0];
  const float* fdst   = (const float*)d_in[1];
  const float* W      = (const float*)d_in[2];
  const float* attn_l = (const float*)d_in[3];
  const float* attn_r = (const float*)d_in[4];
  const float* bias   = (const float*)d_in[5];
  const int*   esg    = (const int*)d_in[6];
  const int*   edg    = (const int*)d_in[7];

  const int n_src   = in_sizes[0] / DIN;
  const int n_dst   = in_sizes[1] / DIN;
  const int nE      = in_sizes[6];
  const int n_total = n_src + n_dst;

  // workspace carve-up (f32): h | el | er | m | z
  float* ws = (float*)d_ws;
  float* h  = ws;
  size_t off = (size_t)n_total * HHD;
  float* el = ws + off;  off += (size_t)n_total * NH;
  float* er = ws + off;  off += (size_t)n_total * NH;
  float* m  = ws + off;  off += (size_t)n_dst * NH;
  float* z  = ws + off;
  float* out = (float*)d_out;

  const dim3 blk(256);

  gat_gemm_wmma<<<dim3((n_total + 127) / 128), blk, 0, stream>>>(
      fsrc, fdst, W, h, n_src, n_total);

  const int t_elr = n_total * NH;
  gat_elr<<<dim3((t_elr + 255) / 256), blk, 0, stream>>>(h, attn_l, attn_r, el, er, n_total);

  const int t_mi = n_dst * NH;
  gat_minit<<<dim3((t_mi + 255) / 256), blk, 0, stream>>>(el, er, m, n_src, n_dst);

  const long long t_em = (long long)nE * NH;
  gat_emax<<<dim3((unsigned)((t_em + 255) / 256)), blk, 0, stream>>>(esg, edg, el, er, m, n_src, nE);

  const int t_init = n_dst * HHD;
  gat_init<<<dim3((t_init + 255) / 256), blk, 0, stream>>>(h, el, er, m, z, out, n_src, n_dst);

  const long long t_acc = (long long)nE * 32;
  gat_eacc<<<dim3((unsigned)((t_acc + 255) / 256)), blk, 0, stream>>>(
      esg, edg, h, el, er, m, z, out, n_src, nE);

  gat_final<<<dim3((t_init + 255) / 256), blk, 0, stream>>>(z, bias, out, n_dst);
}
